// DeformableConv_43147241456356
// MI455X (gfx1250) — matrix operand order repaired and checked by reading, not compiled, not run on hardware
//
#include <hip/hip_runtime.h>
#include <cstdint>

// ---------------------------------------------------------------------------
// Deformable conv v2 block for MI455X (gfx1250, wave32, WMMA).
// Kernel 0: w_def f32 -> f16 (workspace)
// Kernel 1: 3x3 offset conv (128->18) + BN + SiLU -> offsets (workspace)
// Kernel 2: fused bilinear sampling (LDS f16 A-tile) + WMMA GEMM
//           M = B*H*W = 25600, N = 128, K = 128*9 = 1152, fp32 accumulate.
// ---------------------------------------------------------------------------

typedef __attribute__((ext_vector_type(16))) _Float16 v16h;
typedef __attribute__((ext_vector_type(8)))  float    v8f;

#define HH   80
#define WW   80
#define HW   6400
#define CIN  128
#define COUT 128
#define KD   1152            // Cin * 9
#define NB   4
#define LDS_STRIDE 1160      // 1152 + 8 halves pad: 580 dwords % 64 banks = 4

// -------------------- kernel 0: weight f32 -> f16 ---------------------------
__global__ __launch_bounds__(256) void convert_w_f16(
    const float* __restrict__ wd, _Float16* __restrict__ wh, int n)
{
    int i = blockIdx.x * 256 + threadIdx.x;
    if (i < n) wh[i] = (_Float16)wd[i];
}

// -------------------- kernel 1: offset conv + BN + SiLU ---------------------
__global__ __launch_bounds__(256) void offset_conv_bn_silu(
    const float* __restrict__ x,    const float* __restrict__ w_off,
    const float* __restrict__ gamma,const float* __restrict__ beta,
    const float* __restrict__ mean, const float* __restrict__ var,
    float* __restrict__ off)
{
    int idx = blockIdx.x * 256 + threadIdx.x;
    if (idx >= NB * 18 * HW) return;
    int b  = idx / (18 * HW);
    int r  = idx % (18 * HW);
    int co = r / HW;
    int hw = r % HW;
    int h = hw / WW, w = hw % WW;

    const float* xb = x     + (size_t)b  * CIN * HW;
    const float* wp = w_off + (size_t)co * CIN * 9;

    float acc = 0.f;
    for (int ci = 0; ci < CIN; ++ci) {
        const float* xc = xb + ci * HW;
        const float* wc = wp + ci * 9;
        #pragma unroll
        for (int ky = 0; ky < 3; ++ky) {
            int yy = h + ky - 1;
            if (yy < 0 || yy >= HH) continue;
            #pragma unroll
            for (int kx = 0; kx < 3; ++kx) {
                int xx = w + kx - 1;
                if (xx < 0 || xx >= WW) continue;
                acc += xc[yy * WW + xx] * wc[ky * 3 + kx];
            }
        }
    }
    float s = gamma[co] * rsqrtf(var[co] + 1e-5f);
    float v = acc * s + (beta[co] - mean[co] * s);
    off[idx] = v / (1.f + expf(-v));          // SiLU
}

// -------------------- kernel 2: fused sample + WMMA GEMM --------------------
// One block: 16 pixels (one image) x all 128 output channels.
// 8 wave32 waves, each wave = one 16x16 WMMA tile, K = 1152 in 36 steps of 32.
__global__ __launch_bounds__(256) void deform_sample_wmma(
    const float*    __restrict__ x,
    const float*    __restrict__ off,
    const _Float16* __restrict__ wh,     // [COUT][KD] f16, K = c*9+kk
    const float*    __restrict__ bias,
    float*          __restrict__ out)
{
    __shared__ __align__(16) _Float16 As[16 * LDS_STRIDE]; // 16 x 1152 (+pad)
    __shared__ float4 sW[16 * 9];   // bilinear weights*mask per (pixel, tap)
    __shared__ int4   sI[16 * 9];   // clamped neighbor indices (y*W+x)

    const int mBase  = blockIdx.x * 16;     // 16 pixels per block; HW%16==0
    const int b      = mBase / HW;
    const int hwBase = mBase % HW;
    const int t      = threadIdx.x;

    // Phase 1: interpolation metadata for 16 pixels x 9 taps
    if (t < 16 * 9) {
        int p  = t / 9, kk = t % 9;
        int hw = hwBase + p;
        int h = hw / WW, wpx = hw % WW;
        const float* offb = off + (size_t)b * 18 * HW;
        float dy = offb[(2 * kk + 0) * HW + hw];
        float dx = offb[(2 * kk + 1) * HW + hw];
        float ys = (float)(h   + kk / 3 - 1) + dy;
        float xs = (float)(wpx + kk % 3 - 1) + dx;
        float fy0 = floorf(ys), fx0 = floorf(xs);
        int iy0 = (int)fy0, ix0 = (int)fx0;
        int iy1 = iy0 + 1,  ix1 = ix0 + 1;
        float wy1 = ys - fy0, wy0 = 1.f - wy1;
        float wx1 = xs - fx0, wx0 = 1.f - wx1;
        float my0 = (iy0 >= 0 && iy0 < HH) ? 1.f : 0.f;
        float my1 = (iy1 >= 0 && iy1 < HH) ? 1.f : 0.f;
        float mx0 = (ix0 >= 0 && ix0 < WW) ? 1.f : 0.f;
        float mx1 = (ix1 >= 0 && ix1 < WW) ? 1.f : 0.f;
        int cy0 = min(max(iy0, 0), HH - 1), cy1 = min(max(iy1, 0), HH - 1);
        int cx0 = min(max(ix0, 0), WW - 1), cx1 = min(max(ix1, 0), WW - 1);
        sW[t] = make_float4(wy0 * wx0 * my0 * mx0, wy0 * wx1 * my0 * mx1,
                            wy1 * wx0 * my1 * mx0, wy1 * wx1 * my1 * mx1);
        sI[t] = make_int4(cy0 * WW + cx0, cy0 * WW + cx1,
                          cy1 * WW + cx0, cy1 * WW + cx1);
    }
    __syncthreads();

    // Phase 2: gather + interpolate A tile into LDS (f16), K-index = c*9+kk
    const float* xb = x + (size_t)b * CIN * HW;
    for (int e = t; e < 16 * 9 * CIN; e += 256) {
        int c  = e & 127;       // channel (contiguous across lanes)
        int pk = e >> 7;        // p*9 + kk
        float4 wv = sW[pk];
        int4   iv = sI[pk];
        const float* xc = xb + c * HW;
        float v = wv.x * xc[iv.x] + wv.y * xc[iv.y] +
                  wv.z * xc[iv.z] + wv.w * xc[iv.w];
        int p = pk / 9, kk = pk - p * 9;
        As[p * LDS_STRIDE + c * 9 + kk] = (_Float16)v;
    }
    __syncthreads();

    // Phase 3: WMMA GEMM, all 32 lanes active (EXEC all-ones requirement)
    const int lane = t & 31;
    const int wave = t >> 5;
    const int coutBase = wave * 16;
    const int mn   = lane & 15;   // row M for A frag, col N for B frag
    const int kgrp = lane >> 4;   // lane-half selects K sub-block

    union HV { uint4 u[2]; v16h h; };
    const _Float16* arow = As + mn * LDS_STRIDE;
    const _Float16* brow = wh + (size_t)(coutBase + mn) * KD;

    v8f acc = {};
    #pragma unroll 4
    for (int kt = 0; kt < KD / 32; ++kt) {
        // A 16x32 f16: lane half 0 -> K = kt*32 + {0..7, 16..23}
        //              lane half 1 -> K = kt*32 + {8..15, 24..31}
        HV a, bf;
        const _Float16* ap = arow + kt * 32 + kgrp * 8;
        a.u[0] = *(const uint4*)(ap);          // K .. K+7
        a.u[1] = *(const uint4*)(ap + 16);     // K+16 .. K+23
        // B 32x16 f16: same K order as A: {0..7,16..23} / {8..15,24..31}
        const uint4* bp = (const uint4*)(brow + kt * 32 + kgrp * 8);
        bf.u[0] = bp[0];
        bf.u[1] = bp[2];
        acc = __builtin_amdgcn_wmma_f32_16x16x32_f16(
            /*neg_a=*/false, a.h, /*neg_b=*/false, bf.h,
            /*c_mod=*/(short)0, acc, /*reuse_a=*/false, /*reuse_b=*/false);
    }

    // D 16x16 f32: lane = N, VGPR r = pixel row (m = r + kgrp*8)
    float bv = bias[coutBase + mn];
    float* op = out + (size_t)b * COUT * HW
                    + (size_t)(coutBase + mn) * HW
                    + hwBase + kgrp * 8;
    #pragma unroll
    for (int r = 0; r < 8; ++r) op[r] = acc[r] + bv;
}

// ---------------------------------------------------------------------------
extern "C" void kernel_launch(void* const* d_in, const int* in_sizes, int n_in,
                              void* d_out, int out_size, void* d_ws, size_t ws_size,
                              hipStream_t stream)
{
    (void)in_sizes; (void)n_in; (void)out_size; (void)ws_size;
    const float* x     = (const float*)d_in[0];
    const float* w_off = (const float*)d_in[1];
    const float* gamma = (const float*)d_in[2];
    const float* beta  = (const float*)d_in[3];
    const float* mean  = (const float*)d_in[4];
    const float* var   = (const float*)d_in[5];
    const float* w_def = (const float*)d_in[6];
    const float* b_def = (const float*)d_in[7];
    float* out = (float*)d_out;

    // workspace: [offsets f32: 4*18*6400] [weights f16: 128*1152]
    float*    offBuf = (float*)d_ws;
    _Float16* whBuf  = (_Float16*)((char*)d_ws +
                                   (size_t)NB * 18 * HW * sizeof(float));

    convert_w_f16<<<(COUT * KD + 255) / 256, 256, 0, stream>>>(
        w_def, whBuf, COUT * KD);
    offset_conv_bn_silu<<<(NB * 18 * HW + 255) / 256, 256, 0, stream>>>(
        x, w_off, gamma, beta, mean, var, offBuf);
    deform_sample_wmma<<<NB * HW / 16, 256, 0, stream>>>(
        x, offBuf, whBuf, b_def, out);
}
